// HybridKernelRegression_65481071405958
// MI455X (gfx1250) — compile-verified, matrix operand order checked by reading, not run on hardware
//
#include <hip/hip_runtime.h>
#include <hip/hip_bf16.h>
#include <math.h>

// Kernel ridge regression on MI455X:
//   K = exp(-g * max(||x||^2+||y||^2-2 x.y, 0))   -> WMMA GEMM + exp epilogue
//   (K + reg I) a = y                              -> fixed-iteration CG (bandwidth bound:
//                                                     256MB/matvec @ 23.3 TB/s ~ 11us/iter)
//   out = K_test @ a                               -> WMMA gram (reusing workspace) + GEMV
//
// Workspace requirement: N*N floats (256 MiB) + ~160 KiB of CG vectors.

#define GAMMA_ 0.03125f
#define REG_   1e-3f
#define DIM_   32
#define NITER_ 768

typedef float    v2f  __attribute__((ext_vector_type(2)));
typedef float    v8f  __attribute__((ext_vector_type(8)));
typedef _Float16 v16h __attribute__((ext_vector_type(16)));

#if defined(__has_builtin)
#if __has_builtin(__builtin_amdgcn_wmma_f32_16x16x4_f32)
#define HAVE_WMMA_F32 1
#endif
#endif

// ---------------------------------------------------------------------------
// Row squared-norms: xx[i] = sum_d X[i][d]^2   (X is [n, 32] row-major)
// ---------------------------------------------------------------------------
__global__ __launch_bounds__(256) void row_norms_kernel(
    const float* __restrict__ X, float* __restrict__ xx, int n) {
  int i = blockIdx.x * blockDim.x + threadIdx.x;
  if (i >= n) return;
  const float* r = X + (size_t)i * DIM_;
  float s = 0.f;
#pragma unroll
  for (int d = 0; d < DIM_; d += 4) {
    float4 v = *(const float4*)(r + d);
    s += v.x * v.x + v.y * v.y + v.z * v.z + v.w * v.w;
  }
  xx[i] = s;
}

// ---------------------------------------------------------------------------
// RBF gram tile kernel: one wave (32 lanes) computes a 16x16 tile of
//   K[r][c] = exp(-g * max(xxA[r] + xxB[c] - 2 * A[r,:].B[c,:], 0)) (+ diagReg on diag)
// A: [nrows,32], B: [ncols,32] row-major; K row-major with leading dim ldK.
// The D=32 dot product is 8 chained fp32 WMMAs (16x16x4) or one f16 WMMA (16x16x32).
// ---------------------------------------------------------------------------
__global__ __launch_bounds__(32) void rbf_gram_wmma_kernel(
    const float* __restrict__ A, const float* __restrict__ B,
    const float* __restrict__ xxA, const float* __restrict__ xxB,
    float* __restrict__ K, int ldK, float diagReg) {
  const int lane = threadIdx.x;   // 0..31 (wave32)
  const int half = lane >> 4;     // 0 or 1
  const int l16  = lane & 15;
  const int row0 = blockIdx.x * 16;
  const int col0 = blockIdx.y * 16;

  v8f c = {0.f, 0.f, 0.f, 0.f, 0.f, 0.f, 0.f, 0.f};

#ifdef HAVE_WMMA_F32
  // fp32 WMMA: A frag = 16x4 (lane l16 = row, half selects K pair), B symmetric.
  const float* arow = A + (size_t)(row0 + l16) * DIM_ + half * 2;
  const float* brow = B + (size_t)(col0 + l16) * DIM_ + half * 2;
#pragma unroll
  for (int kk = 0; kk < DIM_; kk += 4) {
    v2f a, b;
    a.x = arow[kk]; a.y = arow[kk + 1];
    b.x = brow[kk]; b.y = brow[kk + 1];
    c = __builtin_amdgcn_wmma_f32_16x16x4_f32(false, a, false, b,
                                              (short)0, c, false, false);
  }
#else
  // f16 fallback: 16x16x32 covers D=32 in one WMMA.
  // 16-bit A layout: lane<16 holds K={0..7,16..23}, lane>=16 holds K={8..15,24..31}.
  const float* ar = A + (size_t)(row0 + l16) * DIM_;
  const float* br = B + (size_t)(col0 + l16) * DIM_;
  v16h a, b;
#pragma unroll
  for (int i = 0; i < 16; ++i) {
    int k = (i < 8 ? i : i + 8) + half * 8;
    a[i] = (_Float16)ar[k];
    b[i] = (_Float16)br[k];
  }
  c = __builtin_amdgcn_wmma_f32_16x16x32_f16(false, a, false, b,
                                             (short)0, c, false, false);
#endif

  // Epilogue: C/D layout -> lane holds col = l16, rows v + 8*half for VGPR v.
  const int  ccol = col0 + l16;
  const float xb  = xxB[ccol];
#pragma unroll
  for (int v = 0; v < 8; ++v) {
    int r = row0 + v + half * 8;
    float sq = xxA[r] + xb - 2.0f * c[v];
    sq = fmaxf(sq, 0.0f);
    float kv = __expf(-GAMMA_ * sq);
    if (r == ccol) kv += diagReg;   // fold reg*I into the stored matrix
    K[(size_t)r * (size_t)ldK + ccol] = kv;
  }
}

// ---------------------------------------------------------------------------
// Dense matvec: out[row] = M[row,:] . v   (one block per row, b128 streaming)
// ---------------------------------------------------------------------------
__global__ __launch_bounds__(256) void matvec_kernel(
    const float* __restrict__ Mx, const float* __restrict__ v,
    float* __restrict__ out, int ncols) {
  __shared__ float red[256];
  const int tid = threadIdx.x;
  const float* rp = Mx + (size_t)blockIdx.x * (size_t)ncols;
  float s = 0.f;
  for (int cidx = tid * 4; cidx < ncols; cidx += 256 * 4) {
    float4 k4 = *(const float4*)(rp + cidx);
    float4 v4 = *(const float4*)(v + cidx);
    s += k4.x * v4.x + k4.y * v4.y + k4.z * v4.z + k4.w * v4.w;
  }
  red[tid] = s;
  __syncthreads();
  for (int o = 128; o > 0; o >>= 1) {
    if (tid < o) red[tid] += red[tid + o];
    __syncthreads();
  }
  if (tid == 0) out[blockIdx.x] = red[0];
}

// ---------------------------------------------------------------------------
// CG init: x=0, r=p=y, rho = y.y
// ---------------------------------------------------------------------------
__global__ __launch_bounds__(1024) void cg_init_kernel(
    const float* __restrict__ y, float* __restrict__ x, float* __restrict__ r,
    float* __restrict__ p, float* __restrict__ rho, int n) {
  __shared__ float red[1024];
  const int tid = threadIdx.x;
  float s = 0.f;
  for (int i = tid; i < n; i += 1024) {
    float yv = y[i];
    x[i] = 0.f; r[i] = yv; p[i] = yv;
    s += yv * yv;
  }
  red[tid] = s;
  __syncthreads();
  for (int o = 512; o > 0; o >>= 1) {
    if (tid < o) red[tid] += red[tid + o];
    __syncthreads();
  }
  if (tid == 0) *rho = red[0];
}

// ---------------------------------------------------------------------------
// One CG step (everything after the matvec), single block, device-side scalars:
//   pAp = p.Ap ; a = rho/pAp ; x += a p ; r -= a Ap ; rho' = r.r ;
//   beta = rho'/rho ; p = r + beta p ; rho = rho'
// ---------------------------------------------------------------------------
__global__ __launch_bounds__(1024) void cg_step_kernel(
    float* __restrict__ x, float* __restrict__ r, float* __restrict__ p,
    const float* __restrict__ Ap, float* __restrict__ rho, int n) {
  __shared__ float red[1024];
  const int tid = threadIdx.x;

  float s = 0.f;
  for (int i = tid; i < n; i += 1024) s += p[i] * Ap[i];
  red[tid] = s;
  __syncthreads();
  for (int o = 512; o > 0; o >>= 1) {
    if (tid < o) red[tid] += red[tid + o];
    __syncthreads();
  }
  const float rho_old = *rho;
  const float pAp = red[0];
  const float a = (pAp != 0.f) ? (rho_old / pAp) : 0.f;
  __syncthreads();

  float s2 = 0.f;
  for (int i = tid; i < n; i += 1024) {
    x[i] += a * p[i];
    float rn = r[i] - a * Ap[i];
    r[i] = rn;
    s2 += rn * rn;
  }
  red[tid] = s2;
  __syncthreads();
  for (int o = 512; o > 0; o >>= 1) {
    if (tid < o) red[tid] += red[tid + o];
    __syncthreads();
  }
  const float rho_new = red[0];
  const float beta = (rho_old != 0.f) ? (rho_new / rho_old) : 0.f;
  for (int i = tid; i < n; i += 1024) p[i] = r[i] + beta * p[i];
  if (tid == 0) *rho = rho_new;
}

// ---------------------------------------------------------------------------
extern "C" void kernel_launch(void* const* d_in, const int* in_sizes, int n_in,
                              void* d_out, int out_size, void* d_ws, size_t ws_size,
                              hipStream_t stream) {
  const float* Xtr = (const float*)d_in[0];   // [N, 32]
  const float* ytr = (const float*)d_in[1];   // [N]
  const float* Xte = (const float*)d_in[2];   // [M, 32]

  const int N = in_sizes[1];
  const int M = in_sizes[2] / DIM_;

  float* ws = (float*)d_ws;
  float* Kmat = ws;                               // N*N (reused as M*N for K_test)
  size_t off = (size_t)N * (size_t)N;
  float* xx   = ws + off; off += (size_t)N;       // train row norms
  float* xxt  = ws + off; off += (size_t)M;       // test row norms
  float* xv   = ws + off; off += (size_t)N;       // alpha
  float* rv   = ws + off; off += (size_t)N;       // residual
  float* pv   = ws + off; off += (size_t)N;       // search dir
  float* Apv  = ws + off; off += (size_t)N;       // A*p
  float* rho  = ws + off; off += 1;               // scalar

  // 1) row norms (inputs ~1MB each: L2 resident thereafter)
  row_norms_kernel<<<(N + 255) / 256, 256, 0, stream>>>(Xtr, xx, N);
  row_norms_kernel<<<(M + 255) / 256, 256, 0, stream>>>(Xte, xxt, M);

  // 2) A = K(Xtr, Xtr) + REG*I via WMMA gram
  dim3 g_train((unsigned)(N / 16), (unsigned)(N / 16));
  rbf_gram_wmma_kernel<<<g_train, 32, 0, stream>>>(Xtr, Xtr, xx, xx, Kmat, N, REG_);

  // 3) CG solve A alpha = y (fixed iterations, fully device-side -> capture safe)
  cg_init_kernel<<<1, 1024, 0, stream>>>(ytr, xv, rv, pv, rho, N);
  for (int it = 0; it < NITER_; ++it) {
    matvec_kernel<<<N, 256, 0, stream>>>(Kmat, pv, Apv, N);
    cg_step_kernel<<<1, 1024, 0, stream>>>(xv, rv, pv, Apv, rho, N);
  }

  // 4) K_test = K(Xte, Xtr) reusing the same workspace, then out = K_test @ alpha
  dim3 g_test((unsigned)(M / 16), (unsigned)(N / 16));
  rbf_gram_wmma_kernel<<<g_test, 32, 0, stream>>>(Xte, Xtr, xxt, xx, Kmat, N, 0.0f);
  matvec_kernel<<<M, 256, 0, stream>>>(Kmat, xv, (float*)d_out, N);
}
